// GraphNPE_22093311771387
// MI455X (gfx1250) — matrix operand order repaired and checked by reading, not compiled, not run on hardware
//
#include <hip/hip_runtime.h>

#define NN      262144            // nodes
#define NE      2097152           // edges (no self loops)
#define ET      (NE + NN)         // edges incl. self loops
#define NB      64                // graphs
#define F_IN    8
#define HID     64
#define HEADS   4
#define CH      16                // channels per head
#define LAYERS  3
#define EMB     8
#define SLOPE   0.2f

typedef __attribute__((ext_vector_type(16))) _Float16 v16h;
typedef __attribute__((ext_vector_type(8)))  float    v8f;

// ---- ordered-uint encoding of float (monotonic, handles negatives) ----
__device__ __forceinline__ unsigned fenc(float f) {
  unsigned u = __float_as_uint(f);
  return (u & 0x80000000u) ? ~u : (u | 0x80000000u);
}
__device__ __forceinline__ float fdec(unsigned e) {
  unsigned b = (e & 0x80000000u) ? (e ^ 0x80000000u) : ~e;
  return __uint_as_float(b);
}

// ---------------- utility kernels ----------------
__global__ void k_clear_f32(float* __restrict__ p, int n, float v) {
  int i = blockIdx.x * blockDim.x + threadIdx.x;
  if (i < n) p[i] = v;
}
__global__ void k_clear_u32(unsigned* __restrict__ p, int n, unsigned v) {
  int i = blockIdx.x * blockDim.x + threadIdx.x;
  if (i < n) p[i] = v;
}
// transpose + convert one 64x64 layer weight: Wt[n,k] = (f16) W[k,n]
__global__ void k_w16t(const float* __restrict__ W, _Float16* __restrict__ Wt) {
  int idx = blockIdx.x * blockDim.x + threadIdx.x;   // HID*HID threads
  if (idx >= HID * HID) return;
  int n = idx >> 6, k = idx & 63;
  Wt[n * HID + k] = (_Float16)W[k * HID + n];
}
// fold attention vectors into the weight: Wat[col,k] (16x64, transposed f16)
//   col 0..3  : ws[head=col][k]  = sum_c W[k, col*16+c]  * att_s[col,c]
//   col 4..7  : wd[head=col-4][k]= sum_c W[k, (col-4)*16+c] * att_d[col-4,c]
//   col 8..15 : zero (padding to a full 16-wide WMMA tile)
__global__ void k_watt(const float* __restrict__ W, const float* __restrict__ att_s,
                       const float* __restrict__ att_d, _Float16* __restrict__ Wat) {
  int idx = blockIdx.x * blockDim.x + threadIdx.x;   // 16*HID threads
  if (idx >= 16 * HID) return;
  int col = idx >> 6, k = idx & 63;
  float v = 0.f;
  if (col < 8) {
    int hd = col & 3;
    const float* att = (col < 4) ? att_s : att_d;
#pragma unroll
    for (int c = 0; c < CH; ++c) v += W[k * HID + hd * CH + c] * att[hd * CH + c];
  }
  Wat[col * HID + k] = (_Float16)v;
}

// ---------------- input projection: h = relu(x @ W_in + b_in); also emit f16 copy ----
__global__ void k_input_proj(const float* __restrict__ x, const float* __restrict__ W,
                             const float* __restrict__ b, float* __restrict__ h,
                             _Float16* __restrict__ hh) {
  int idx = blockIdx.x * blockDim.x + threadIdx.x;   // NN*HID threads
  if (idx >= NN * HID) return;
  int n = idx >> 6, j = idx & 63;
  const float* xr = x + (size_t)n * F_IN;
  float acc = b[j];
#pragma unroll
  for (int f = 0; f < F_IN; ++f) acc += xr[f] * W[f * HID + j];
  acc = fmaxf(acc, 0.f);
  h[idx]  = acc;
  hh[idx] = (_Float16)acc;
}

// ---------------- WMMA GEMM + attention logits as extra column tile ----------------
// per 16-row tile: tiles 0..3 -> hp[.,16 cols of head t]; tile 4 -> [al_s | al_d] cols.
// K=64 -> 2x v_wmma_f32_16x16x32_f16 per tile.
__global__ __launch_bounds__(256)
void k_gemm_gat(const _Float16* __restrict__ A, const _Float16* __restrict__ Wt,
                const _Float16* __restrict__ Wat, float* __restrict__ out,
                float* __restrict__ al_s, float* __restrict__ al_d) {
  __shared__ __align__(16) _Float16 Ws[HID * HID];   // Wt staged in LDS, 8 KB
  __shared__ __align__(16) _Float16 Was[16 * HID];   // folded att weights, 2 KB
  for (int i = threadIdx.x; i < (HID * HID) / 8; i += 256)
    ((uint4*)Ws)[i] = ((const uint4*)Wt)[i];
  for (int i = threadIdx.x; i < (16 * HID) / 8; i += 256)
    ((uint4*)Was)[i] = ((const uint4*)Wat)[i];
  __syncthreads();

  const int lane = threadIdx.x & 31;
  const int wid  = (blockIdx.x * 256 + threadIdx.x) >> 5;   // 81920 waves total
  const int row0 = (wid / 5) * 16;                          // 16-row tile
  const int t    = wid % 5;                                 // 0..3 = head tile, 4 = logits

  const int m    = lane & 15;
  const int half = lane >> 4;

  const _Float16* arow = A + (size_t)(row0 + m) * HID;
  __builtin_prefetch(arow + (size_t)16 * HID, 0, 1);        // next row tile

  // wave-uniform B base (EXEC stays full through the WMMAs)
  const _Float16* Bbase = (t < 4) ? (Ws + (t * CH + m) * HID) : (Was + m * HID);

  v8f acc = {};
#pragma unroll
  for (int k0 = 0; k0 < HID; k0 += 32) {
    v16h a, b;
    // A 16x32 f16: lanes 0-15 hold K=k0+0..15; lanes 16-31 K=k0+16..31
#pragma unroll
    for (int i = 0; i < 8; ++i) {
      a[i]     = arow[k0 + half * 16 + i];
      a[8 + i] = arow[k0 + half * 16 + 8 + i];
    }
    // B 32x16 f16: lane = col twice over; lanes 0-15 K=k0+0..15, lanes 16-31 K=k0+16..31.
    // B sources are [col,k]-major so each lane reads 16 contiguous halves (2x ds_load_b128).
    const _Float16* brow = Bbase + k0 + half * 16;
#pragma unroll
    for (int i = 0; i < 16; ++i) b[i] = brow[i];

    acc = __builtin_amdgcn_wmma_f32_16x16x32_f16(false, a, false, b,
                                                 (short)0, acc, false, false);
  }

  // D layout: VGPR r -> M = r + half*8, N = m
  if (t < 4) {
#pragma unroll
    for (int r = 0; r < 8; ++r)
      out[(size_t)(row0 + half * 8 + r) * HID + t * CH + m] = acc[r];
  } else if (m < 8) {
    float* dst = (m < 4) ? al_s : al_d;
    int hd = m & 3;
#pragma unroll
    for (int r = 0; r < 8; ++r)
      dst[(row0 + half * 8 + r) * HEADS + hd] = acc[r];
  }
}

// ---------------- edge pass 1: segment max (ordered-uint atomic max) ----------------
__global__ void k_edge_max(const int* __restrict__ ei, const float* __restrict__ al_s,
                           const float* __restrict__ al_d, unsigned* __restrict__ mkey) {
  int e = blockIdx.x * blockDim.x + threadIdx.x;
  if (e >= ET) return;
  int s, d;
  if (e < NE) { s = ei[e]; d = ei[NE + e]; } else { s = d = e - NE; }
#pragma unroll
  for (int h = 0; h < HEADS; ++h) {
    float v = al_s[s * HEADS + h] + al_d[d * HEADS + h];
    v = (v > 0.f) ? v : SLOPE * v;
    atomicMax(&mkey[d * HEADS + h], fenc(v));
  }
}

// ---------------- edge pass 2: ex = exp(e - max); denom += ex ----------------
__global__ void k_edge_exp(const int* __restrict__ ei, const float* __restrict__ al_s,
                           const float* __restrict__ al_d, const unsigned* __restrict__ mkey,
                           float* __restrict__ exs, float* __restrict__ denom) {
  int e = blockIdx.x * blockDim.x + threadIdx.x;
  if (e >= ET) return;
  int s, d;
  if (e < NE) { s = ei[e]; d = ei[NE + e]; } else { s = d = e - NE; }
#pragma unroll
  for (int h = 0; h < HEADS; ++h) {
    float v = al_s[s * HEADS + h] + al_d[d * HEADS + h];
    v = (v > 0.f) ? v : SLOPE * v;
    float ex = __expf(v - fdec(mkey[d * HEADS + h]));
    exs[(size_t)e * HEADS + h] = ex;
    atomicAdd(&denom[d * HEADS + h], ex);
  }
}

// ---------------- edge pass 3: agg[dst] += alpha * hp[src] ----------------
__global__ void k_edge_scatter(const int* __restrict__ ei, const float* __restrict__ exs,
                               const float* __restrict__ denom, const float* __restrict__ hp,
                               float* __restrict__ agg) {
  int t = blockIdx.x * blockDim.x + threadIdx.x;     // ET*HEADS threads
  if (t >= ET * HEADS) return;
  int e = t >> 2, h = t & 3;
  int s, d;
  if (e < NE) { s = ei[e]; d = ei[NE + e]; } else { s = d = e - NE; }
  float alpha = exs[t] / (denom[d * HEADS + h] + 1e-16f);
  const float* srow = hp  + (size_t)s * HID + h * CH;
  float*       drow = agg + (size_t)d * HID + h * CH;
#pragma unroll
  for (int c = 0; c < CH; ++c) atomicAdd(&drow[c], srow[c] * alpha);
}

// ---------------- bias + relu; also emit f16 copy for next layer's WMMA ----------------
__global__ void k_bias_relu(float* __restrict__ h, _Float16* __restrict__ hh,
                            const float* __restrict__ b) {
  int idx = blockIdx.x * blockDim.x + threadIdx.x;   // NN*HID threads
  if (idx >= NN * HID) return;
  float v = fmaxf(h[idx] + b[idx & 63], 0.f);
  h[idx]  = v;
  hh[idx] = (_Float16)v;
}

// ---------------- global mean pool (sums + counts) ----------------
__global__ void k_pool(const float* __restrict__ h, const int* __restrict__ batch,
                       float* __restrict__ sums, float* __restrict__ cnts) {
  int idx = blockIdx.x * blockDim.x + threadIdx.x;   // NN*HID threads
  if (idx >= NN * HID) return;
  int n = idx >> 6, j = idx & 63;
  int b = batch[n];
  atomicAdd(&sums[b * HID + j], h[idx]);
  if (j == 0) atomicAdd(&cnts[b], 1.f);
}

// ---------------- tiny output MLP: [64,64] -> relu -> [64,8] ----------------
__global__ void k_mlp(const float* __restrict__ sums, const float* __restrict__ cnts,
                      const float* __restrict__ W1, const float* __restrict__ b1,
                      const float* __restrict__ W2, const float* __restrict__ b2,
                      float* __restrict__ out) {
  int bb = threadIdx.x;                              // <<<1,64>>>
  if (bb >= NB) return;
  float inv = 1.f / fmaxf(cnts[bb], 1.f);
  float p[HID], t[HID];
#pragma unroll
  for (int j = 0; j < HID; ++j) p[j] = sums[bb * HID + j] * inv;
  for (int j = 0; j < HID; ++j) {
    float a = b1[j];
#pragma unroll
    for (int k = 0; k < HID; ++k) a += p[k] * W1[k * HID + j];
    t[j] = fmaxf(a, 0.f);
  }
  for (int o = 0; o < EMB; ++o) {
    float a = b2[o];
#pragma unroll
    for (int j = 0; j < HID; ++j) a += t[j] * W2[j * EMB + o];
    out[bb * EMB + o] = a;
  }
}

// =======================================================================
extern "C" void kernel_launch(void* const* d_in, const int* in_sizes, int n_in,
                              void* d_out, int out_size, void* d_ws, size_t ws_size,
                              hipStream_t stream) {
  const float* x       = (const float*)d_in[0];
  const int*   ei      = (const int*)  d_in[1];
  const int*   batch   = (const int*)  d_in[2];
  const float* W_in    = (const float*)d_in[3];
  const float* b_in    = (const float*)d_in[4];
  const float* W_gat   = (const float*)d_in[5];
  const float* att_src = (const float*)d_in[6];
  const float* att_dst = (const float*)d_in[7];
  const float* b_gat   = (const float*)d_in[8];
  const float* W1      = (const float*)d_in[9];
  const float* b1      = (const float*)d_in[10];
  const float* W2      = (const float*)d_in[11];
  const float* b2      = (const float*)d_in[12];
  float* out = (float*)d_out;

  // ---- workspace carve-out ----
  char* ws = (char*)d_ws;
  size_t off = 0;
  auto carve = [&](size_t bytes) -> void* {
    void* p = ws + off;
    off = (off + bytes + 255) & ~(size_t)255;
    return p;
  };
  float*    h     = (float*)   carve((size_t)NN * HID * 4);   // node features / agg accumulator
  float*    hp    = (float*)   carve((size_t)NN * HID * 4);   // W-projected features
  _Float16* hh    = (_Float16*)carve((size_t)NN * HID * 2);   // f16 copy for WMMA
  _Float16* Wt    = (_Float16*)carve((size_t)HID * HID * 2);  // transposed f16 layer weight
  _Float16* Wat   = (_Float16*)carve((size_t)16 * HID * 2);   // folded attention weights
  float*    al_s  = (float*)   carve((size_t)NN * HEADS * 4);
  float*    al_d  = (float*)   carve((size_t)NN * HEADS * 4);
  unsigned* mkey  = (unsigned*)carve((size_t)NN * HEADS * 4);
  float*    denom = (float*)   carve((size_t)NN * HEADS * 4);
  float*    exs   = (float*)   carve((size_t)ET * HEADS * 4);
  float*    sums  = (float*)   carve((size_t)NB * HID * 4);
  float*    cnts  = (float*)   carve((size_t)NB * 4);

  const int TPB = 256;
  const int nh_blocks  = (NN * HID + TPB - 1) / TPB;          // 65536
  const int nh4_blocks = (NN * HEADS + TPB - 1) / TPB;
  const int e_blocks   = (ET + TPB - 1) / TPB;
  const int eh_blocks  = (ET * HEADS + TPB - 1) / TPB;
  const int gemm_blocks = (NN / 16) * 5 / 8;                  // 5 tiles/row, 8 waves/block

  // input projection (writes f32 + f16 activations)
  k_input_proj<<<nh_blocks, TPB, 0, stream>>>(x, W_in, b_in, h, hh);

  for (int l = 0; l < LAYERS; ++l) {
    // transposed f16 layer weight + folded attention columns
    k_w16t<<<(HID * HID + TPB - 1) / TPB, TPB, 0, stream>>>(
        W_gat + (size_t)l * HID * HID, Wt);
    k_watt<<<(16 * HID + TPB - 1) / TPB, TPB, 0, stream>>>(
        W_gat + (size_t)l * HID * HID, att_src + l * HEADS * CH,
        att_dst + l * HEADS * CH, Wat);

    // WMMA GEMM producing hp AND the per-node attention logits
    k_gemm_gat<<<gemm_blocks, TPB, 0, stream>>>(hh, Wt, Wat, hp, al_s, al_d);

    // reset accumulators (h becomes the aggregation target)
    k_clear_f32<<<nh_blocks, TPB, 0, stream>>>(h, NN * HID, 0.f);
    k_clear_u32<<<nh4_blocks, TPB, 0, stream>>>(mkey, NN * HEADS, 0u);
    k_clear_f32<<<nh4_blocks, TPB, 0, stream>>>(denom, NN * HEADS, 0.f);

    // segment softmax over incoming edges + weighted scatter
    k_edge_max    <<<e_blocks,  TPB, 0, stream>>>(ei, al_s, al_d, mkey);
    k_edge_exp    <<<e_blocks,  TPB, 0, stream>>>(ei, al_s, al_d, mkey, exs, denom);
    k_edge_scatter<<<eh_blocks, TPB, 0, stream>>>(ei, exs, denom, hp, h);

    // bias + relu (also produce next layer's f16 activations)
    k_bias_relu<<<nh_blocks, TPB, 0, stream>>>(h, hh, b_gat + (size_t)l * HID);
  }

  // global mean pool + output MLP
  k_clear_f32<<<(NB * HID + TPB - 1) / TPB, TPB, 0, stream>>>(sums, NB * HID, 0.f);
  k_clear_f32<<<1, NB, 0, stream>>>(cnts, NB, 0.f);
  k_pool<<<nh_blocks, TPB, 0, stream>>>(h, batch, sums, cnts);
  k_mlp<<<1, 64, 0, stream>>>(sums, cnts, W1, b1, W2, b2, out);
}
